// MultiHeadAttention_44074954392148
// MI455X (gfx1250) — matrix operand order repaired and checked by reading, not compiled, not run on hardware
//
#include <hip/hip_runtime.h>

// ---------------------------------------------------------------------------
// Fused causal MHA for MI455X (gfx1250): bf16 WMMA + f32 accumulate,
// TDM (tensor_load_to_lds) tile staging, double-buffered software pipeline
// (1 barrier/iter, TDM + global loads overlap WMMA), reuse-A WMMA hints.
// B=4, S=2048, E=1024, H=16, D=64.
// ---------------------------------------------------------------------------

typedef __bf16 bf16;
typedef __attribute__((ext_vector_type(16))) __bf16 v16bf;
typedef __attribute__((ext_vector_type(8)))  __bf16 v8bf;
typedef __attribute__((ext_vector_type(8)))  float  v8f;
typedef __attribute__((ext_vector_type(4)))  float  v4f;

constexpr int Bn = 4, Sn = 2048, En = 1024, Hn = 16, Dn = 64;
constexpr int BS = Bn * Sn;

#define DEV static __device__ __forceinline__

#if defined(__has_builtin)
#if __has_builtin(__builtin_amdgcn_tensor_load_to_lds)
#define HAVE_TDM 1
#endif
#endif
#ifndef HAVE_TDM
#define HAVE_TDM 0
#endif

DEV v16bf mk16(v8bf lo, v8bf hi) {
  return __builtin_shufflevector(lo, hi, 0,1,2,3,4,5,6,7,8,9,10,11,12,13,14,15);
}
DEV v8f zero8() { v8f z = {0.f,0.f,0.f,0.f,0.f,0.f,0.f,0.f}; return z; }
DEV v8f wmma_bf16(v16bf a, v16bf b, v8f c) {
  return __builtin_amdgcn_wmma_f32_16x16x32_bf16(false, a, false, b, (short)0, c,
                                                 false, false);
}
// reuse-A hint: only legal when previous instruction is the identical WMMA.
DEV v8f wmma_bf16_ra(v16bf a, v16bf b, v8f c) {
  return __builtin_amdgcn_wmma_f32_16x16x32_bf16(false, a, false, b, (short)0, c,
                                                 true, false);
}
DEV float rmax16(float v) {
#pragma unroll
  for (int m = 1; m < 16; m <<= 1) v = fmaxf(v, __shfl_xor(v, m, 32));
  return v;
}
DEV float rsum16(float v) {
#pragma unroll
  for (int m = 1; m < 16; m <<= 1) v += __shfl_xor(v, m, 32);
  return v;
}

// ---------------------------------------------------------------------------
// Tensor Data Mover: 2D bf16 tile Global -> LDS with LDS row padding.
// D# per CDNA5 ISA 8.3/8.4. 6-arg builtin:
//   (u32x4 g0, i32x8 g1, i32x4 g2, i32x4 g3, i32x8, i32 cpol)
// ---------------------------------------------------------------------------
#if HAVE_TDM
typedef __attribute__((ext_vector_type(4))) unsigned int v4u;
typedef __attribute__((ext_vector_type(8))) int v8i;
typedef __attribute__((ext_vector_type(4))) int v4i;

DEV unsigned lds_off(const void* p) {
  return (unsigned)(unsigned long long)(size_t)p;  // low 32 bits = LDS offset
}

DEV void tdm_load_tile_bf16(unsigned lds_byte_addr, const bf16* gsrc,
                            int tile_w, int tile_h, int row_stride_elems,
                            int pad_interval_code, int pad_amount_code) {
  unsigned long long ga = (unsigned long long)(size_t)gsrc;
  v4u g0;
  g0[0] = 1u;                                    // count=1, user descriptor
  g0[1] = lds_byte_addr;                         // lds_addr
  g0[2] = (unsigned)ga;                          // global_addr[31:0]
  g0[3] = (unsigned)((ga >> 32) & 0x01ffffffull) // global_addr[56:32]
          | (2u << 30);                          // type=2 ("image")
  v8i g1;
  g1[0] = (int)((1u << 16)                       // data_size = 2 bytes
                | (1u << 20)                     // pad_enable
                | ((unsigned)pad_interval_code << 22)
                | ((unsigned)pad_amount_code << 25));
  g1[1] = (int)(((unsigned)tile_w & 0xffffu) << 16);  // tensor_dim0[15:0]
  g1[2] = (int)(((unsigned)tile_h & 0xffffu) << 16);  // tensor_dim1[15:0]
  g1[3] = (int)(((unsigned)tile_w & 0xffffu) << 16);  // tile_dim0
  g1[4] = (int)((unsigned)tile_h & 0xffffu);          // tile_dim1 (tile_dim2=0)
  g1[5] = row_stride_elems;                           // tensor_dim0_stride
  g1[6] = 0;
  g1[7] = 0;
  v4i z4 = {0, 0, 0, 0};
  v8i z8 = {0, 0, 0, 0, 0, 0, 0, 0};
  __builtin_amdgcn_tensor_load_to_lds(g0, g1, z4, z4, z8, 0);
}

DEV void tdm_wait0() {
#if __has_builtin(__builtin_amdgcn_s_wait_tensorcnt)
  __builtin_amdgcn_s_wait_tensorcnt(0);
#else
  asm volatile("s_wait_tensorcnt 0x0" ::: "memory");
#endif
}
#endif  // HAVE_TDM

// ---------------------------------------------------------------------------
// f32 -> bf16 conversion, 8 elems/thread (all sizes are multiples of 8)
// ---------------------------------------------------------------------------
__global__ __launch_bounds__(256) void cvt_k(const float* __restrict__ in,
                                             bf16* __restrict__ out, int n) {
  int i = (blockIdx.x * blockDim.x + threadIdx.x) * 8;
  if (i < n) {
    v4f f0 = *(const v4f*)&in[i];
    v4f f1 = *(const v4f*)&in[i + 4];
    v8bf o;
#pragma unroll
    for (int j = 0; j < 4; j++) { o[j] = (bf16)f0[j]; o[4 + j] = (bf16)f1[j]; }
    *(v8bf*)&out[i] = o;
  }
}

// ---------------------------------------------------------------------------
// QKV projection: per (mat, head) GEMM (BSxE) x (ExD) -> (B,H,S,D) bf16.
// 128x64 tile, K-chunks of 32, 8 waves, double-buffered pipeline.
// ---------------------------------------------------------------------------
__global__ __launch_bounds__(256) void qkv_gemm_k(
    const bf16* __restrict__ X, const bf16* __restrict__ Wq,
    const bf16* __restrict__ Wk, const bf16* __restrict__ Wv,
    bf16* __restrict__ Qo, bf16* __restrict__ Ko, bf16* __restrict__ Vo) {
  __shared__ bf16 lA[2][128][40];  // [m][k]: 64B rows + 16B pad
  __shared__ bf16 lB[2][64][40];   // W tile transposed: [n][k]

  const int mt = blockIdx.x;
  const int z  = blockIdx.y;          // mat*16 + head
  const int mat = z >> 4, h = z & 15;
  const bf16* __restrict__ W =
      (mat == 0 ? Wq : (mat == 1 ? Wk : Wv)) + (size_t)h * En * Dn;
  bf16* __restrict__ Out = (mat == 0 ? Qo : (mat == 1 ? Ko : Vo));

  const int row0 = mt * 128;
  const int b = row0 / Sn;
  const int s0 = row0 - b * Sn;
  const int tid = threadIdx.x;
  const int wave = tid >> 5, lane = tid & 31;
  const int l16 = lane & 15, lh = lane >> 4;
  const int kb = lh * 8, kb2 = lh * 16;

  v8f acc[4];
#pragma unroll
  for (int t = 0; t < 4; t++) acc[t] = zero8();

  const int bk = tid >> 3, bn = (tid & 7) * 8;        // B coop load
#if !HAVE_TDM
  const int arow = tid >> 1, ahalf = (tid & 1) * 16;  // A coop load fallback
#endif

  // ---- prologue: stage tile 0 ----
  v8bf w = *(const v8bf*)&W[(size_t)bk * Dn + bn];
#if HAVE_TDM
  if (wave == 0)
    tdm_load_tile_bf16(lds_off(&lA[0][0][0]), &X[(size_t)row0 * En],
                       32, 128, En, /*2^3 x8B = 64B*/ 3, /*4 DW = 16B*/ 3);
#else
  v8bf a0 = *(const v8bf*)&X[(size_t)(row0 + arow) * En + ahalf];
  v8bf a1 = *(const v8bf*)&X[(size_t)(row0 + arow) * En + ahalf + 8];
#endif

  const int niter = En / 32;
  for (int i = 0; i < niter; i++) {
    const int cur = i & 1, nxt = cur ^ 1;
    const int k0 = i * 32;

    // publish register-staged tiles into LDS[cur]
#pragma unroll
    for (int j = 0; j < 8; j++) lB[cur][bn + j][bk] = w[j];
#if !HAVE_TDM
    *(v8bf*)&lA[cur][arow][ahalf]     = a0;
    *(v8bf*)&lA[cur][arow][ahalf + 8] = a1;
#endif
    // issue next tile's global loads (latency covered by compute below)
    if (i + 1 < niter) {
      w = *(const v8bf*)&W[(size_t)(k0 + 32 + bk) * Dn + bn];
#if !HAVE_TDM
      a0 = *(const v8bf*)&X[(size_t)(row0 + arow) * En + k0 + 32 + ahalf];
      a1 = *(const v8bf*)&X[(size_t)(row0 + arow) * En + k0 + 32 + ahalf + 8];
#endif
    }
#if HAVE_TDM
    if (wave == 0) tdm_wait0();          // TDM(i) complete
#endif
    __syncthreads();                     // publish lA/lB[cur]
#if HAVE_TDM
    // safe after barrier: all reads of lA[nxt] (iter i-1) have completed
    if (i + 1 < niter && wave == 0)
      tdm_load_tile_bf16(lds_off(&lA[nxt][0][0]),
                         &X[(size_t)row0 * En + k0 + 32], 32, 128, En, 3, 3);
#endif

    const int frow = wave * 16 + l16;
    v16bf a = mk16(*(const v8bf*)&lA[cur][frow][kb],
                   *(const v8bf*)&lA[cur][frow][kb + 16]);
    v16bf bb[4];
#pragma unroll
    for (int t = 0; t < 4; t++) {
      const int col = t * 16 + l16;
      bb[t] = mk16(*(const v8bf*)&lB[cur][col][kb],
                   *(const v8bf*)&lB[cur][col][kb + 16]);
    }
    acc[0] = wmma_bf16(a, bb[0], acc[0]);
#pragma unroll
    for (int t = 1; t < 4; t++) acc[t] = wmma_bf16_ra(a, bb[t], acc[t]);
    // no trailing barrier: iter i+1 writes the other buffer; iter i+2 writes
    // this buffer only after barrier(i+1), past which all reads here are done.
  }

#pragma unroll
  for (int t = 0; t < 4; t++) {
    const int n = t * 16 + l16;
#pragma unroll
    for (int r = 0; r < 8; r++) {
      const int s = s0 + wave * 16 + r + 8 * lh;
      Out[(((size_t)b * Hn + h) * Sn + s) * Dn + n] = (bf16)acc[t][r];
    }
  }
}

// ---------------------------------------------------------------------------
// Flash attention: grid (S/128, H, B), 8 waves; wave w owns 16 q rows.
// K tile via TDM, V tile register-staged + transposed; double-buffered.
// ---------------------------------------------------------------------------
__global__ __launch_bounds__(256) void attn_k(
    const bf16* __restrict__ Q, const bf16* __restrict__ K,
    const bf16* __restrict__ V, bf16* __restrict__ Ctx) {
  __shared__ bf16 lK[2][32][72];   // [key][d]: 128B rows + 16B pad
  __shared__ bf16 lVT[2][64][40];  // V transposed: [d][key]
  __shared__ bf16 lP[8][16][32];   // per-wave P scratch [m][key]

  const int qt = blockIdx.x, h = blockIdx.y, b = blockIdx.z;
  const int qbase = qt * 128;
  const int tid = threadIdx.x, wave = tid >> 5, lane = tid & 31;
  const int l16 = lane & 15, lh = lane >> 4;
  const int kb = lh * 8, kb2 = lh * 16;
  const size_t bh = ((size_t)b * Hn + h) * Sn;

  const int qrow = qbase + wave * 16 + l16;
  v16bf qa[2];
#pragma unroll
  for (int i = 0; i < 2; i++) {
    const bf16* src = &Q[(bh + qrow) * Dn + i * 32 + kb];
    qa[i] = mk16(*(const v8bf*)src, *(const v8bf*)(src + 16));
  }

  float mrun[8], lrun[8];
#pragma unroll
  for (int r = 0; r < 8; r++) { mrun[r] = -3.0e38f; lrun[r] = 0.f; }
  v8f o[4];
#pragma unroll
  for (int t = 0; t < 4; t++) o[t] = zero8();

  const int ldk = tid >> 3, ldd = (tid & 7) * 8;
  const int ntiles = qt * 4 + 4;

  // ---- prologue: stage tile 0 ----
  v8bf vv = *(const v8bf*)&V[(bh + ldk) * Dn + ldd];
#if HAVE_TDM
  if (wave == 0)
    tdm_load_tile_bf16(lds_off(&lK[0][0][0]), &K[bh * Dn],
                       64, 32, 64, /*2^4 x8B = 128B*/ 4, /*16B*/ 3);
#else
  v8bf kk = *(const v8bf*)&K[(bh + ldk) * Dn + ldd];
#endif

  for (int j = 0; j < ntiles; j++) {
    const int cur = j & 1, nxt = cur ^ 1;
    const int kbase = j * 32;

    // publish register-staged tiles into LDS[cur]
#pragma unroll
    for (int jj = 0; jj < 8; jj++) lVT[cur][ldd + jj][ldk] = vv[jj];
#if !HAVE_TDM
    *(v8bf*)&lK[cur][ldk][ldd] = kk;
#endif
    if (j + 1 < ntiles) {
      vv = *(const v8bf*)&V[(bh + kbase + 32 + ldk) * Dn + ldd];
#if !HAVE_TDM
      kk = *(const v8bf*)&K[(bh + kbase + 32 + ldk) * Dn + ldd];
#endif
    }
#if HAVE_TDM
    if (wave == 0) tdm_wait0();
#endif
    __syncthreads();
#if HAVE_TDM
    if (j + 1 < ntiles && wave == 0)
      tdm_load_tile_bf16(lds_off(&lK[nxt][0][0]), &K[(bh + kbase + 32) * Dn],
                         64, 32, 64, 4, 3);
#endif

    // scores: S = Q K^T; load all B frags first, paired WMMAs (reuse A)
    v16bf kf0[2], kf1[2];
#pragma unroll
    for (int t = 0; t < 2; t++) {
      const int key = t * 16 + l16;
      kf0[t] = mk16(*(const v8bf*)&lK[cur][key][kb],
                    *(const v8bf*)&lK[cur][key][kb + 16]);
      kf1[t] = mk16(*(const v8bf*)&lK[cur][key][32 + kb],
                    *(const v8bf*)&lK[cur][key][32 + kb + 16]);
    }
    v8f sc[2];
    sc[0] = wmma_bf16(qa[0], kf0[0], zero8());
    sc[1] = wmma_bf16_ra(qa[0], kf0[1], zero8());
    sc[0] = wmma_bf16(qa[1], kf1[0], sc[0]);
    sc[1] = wmma_bf16_ra(qa[1], kf1[1], sc[1]);

    // causal mask + scale + online softmax (row m = r + 8*lh per lane)
    float pv[2][8], corr[8];
#pragma unroll
    for (int r = 0; r < 8; r++) {
      const int qg = qbase + wave * 16 + r + 8 * lh;
      float s0 = (kbase + l16      <= qg) ? sc[0][r] * 0.125f : -3.0e38f;
      float s1 = (kbase + 16 + l16 <= qg) ? sc[1][r] * 0.125f : -3.0e38f;
      float rm = rmax16(fmaxf(s0, s1));
      float nm = fmaxf(mrun[r], rm);
      float p0 = __expf(s0 - nm);
      float p1 = __expf(s1 - nm);
      float rs = rsum16(p0 + p1);
      corr[r] = __expf(mrun[r] - nm);
      lrun[r] = lrun[r] * corr[r] + rs;
      mrun[r] = nm;
      pv[0][r] = p0; pv[1][r] = p1;
    }
#pragma unroll
    for (int t = 0; t < 4; t++)
#pragma unroll
      for (int r = 0; r < 8; r++) o[t][r] *= corr[r];

    // P: C-layout -> per-wave LDS -> A-layout fragment (same-wave, in order)
#pragma unroll
    for (int t = 0; t < 2; t++)
#pragma unroll
      for (int r = 0; r < 8; r++)
        lP[wave][r + 8 * lh][t * 16 + l16] = (bf16)pv[t][r];

    v16bf pa = mk16(*(const v8bf*)&lP[wave][l16][kb],
                    *(const v8bf*)&lP[wave][l16][kb + 16]);
    v16bf bv[4];
#pragma unroll
    for (int t = 0; t < 4; t++) {
      const int dcol = t * 16 + l16;
      bv[t] = mk16(*(const v8bf*)&lVT[cur][dcol][kb],
                   *(const v8bf*)&lVT[cur][dcol][kb + 16]);
    }
    o[0] = wmma_bf16(pa, bv[0], o[0]);
#pragma unroll
    for (int t = 1; t < 4; t++) o[t] = wmma_bf16_ra(pa, bv[t], o[t]);
  }

#pragma unroll
  for (int r = 0; r < 8; r++) {
    const int qg = qbase + wave * 16 + r + 8 * lh;
    const float inv = 1.f / lrun[r];
#pragma unroll
    for (int t = 0; t < 4; t++)
      Ctx[((size_t)b * Sn + qg) * En + h * Dn + t * 16 + l16] =
          (bf16)(o[t][r] * inv);
  }
}

// ---------------------------------------------------------------------------
// Output projection: (BSxE bf16) x (ExE bf16) + bias -> f32 out.
// ---------------------------------------------------------------------------
__global__ __launch_bounds__(256) void proj_k(
    const bf16* __restrict__ A, const bf16* __restrict__ Wp,
    const float* __restrict__ bp, float* __restrict__ Out) {
  __shared__ bf16 lA[2][128][40];
  __shared__ bf16 lB[2][64][40];

  const int mt = blockIdx.x, nt = blockIdx.y;
  const int row0 = mt * 128, n0 = nt * 64;
  const int tid = threadIdx.x, wave = tid >> 5, lane = tid & 31;
  const int l16 = lane & 15, lh = lane >> 4;
  const int kb = lh * 8, kb2 = lh * 16;

  v8f acc[4];
#pragma unroll
  for (int t = 0; t < 4; t++) acc[t] = zero8();

  const int bk = tid >> 3, bn = (tid & 7) * 8;
#if !HAVE_TDM
  const int arow = tid >> 1, ahalf = (tid & 1) * 16;
#endif

  // ---- prologue ----
  v8bf w = *(const v8bf*)&Wp[(size_t)bk * En + n0 + bn];
#if HAVE_TDM
  if (wave == 0)
    tdm_load_tile_bf16(lds_off(&lA[0][0][0]), &A[(size_t)row0 * En],
                       32, 128, En, 3, 3);
#else
  v8bf a0 = *(const v8bf*)&A[(size_t)(row0 + arow) * En + ahalf];
  v8bf a1 = *(const v8bf*)&A[(size_t)(row0 + arow) * En + ahalf + 8];
#endif

  const int niter = En / 32;
  for (int i = 0; i < niter; i++) {
    const int cur = i & 1, nxt = cur ^ 1;
    const int k0 = i * 32;

#pragma unroll
    for (int j = 0; j < 8; j++) lB[cur][bn + j][bk] = w[j];
#if !HAVE_TDM
    *(v8bf*)&lA[cur][arow][ahalf]     = a0;
    *(v8bf*)&lA[cur][arow][ahalf + 8] = a1;
#endif
    if (i + 1 < niter) {
      w = *(const v8bf*)&Wp[(size_t)(k0 + 32 + bk) * En + n0 + bn];
#if !HAVE_TDM
      a0 = *(const v8bf*)&A[(size_t)(row0 + arow) * En + k0 + 32 + ahalf];
      a1 = *(const v8bf*)&A[(size_t)(row0 + arow) * En + k0 + 32 + ahalf + 8];
#endif
    }
#if HAVE_TDM
    if (wave == 0) tdm_wait0();
#endif
    __syncthreads();
#if HAVE_TDM
    if (i + 1 < niter && wave == 0)
      tdm_load_tile_bf16(lds_off(&lA[nxt][0][0]),
                         &A[(size_t)row0 * En + k0 + 32], 32, 128, En, 3, 3);
#endif

    const int frow = wave * 16 + l16;
    v16bf a = mk16(*(const v8bf*)&lA[cur][frow][kb],
                   *(const v8bf*)&lA[cur][frow][kb + 16]);
    v16bf bb[4];
#pragma unroll
    for (int t = 0; t < 4; t++) {
      const int col = t * 16 + l16;
      bb[t] = mk16(*(const v8bf*)&lB[cur][col][kb],
                   *(const v8bf*)&lB[cur][col][kb + 16]);
    }
    acc[0] = wmma_bf16(a, bb[0], acc[0]);
#pragma unroll
    for (int t = 1; t < 4; t++) acc[t] = wmma_bf16_ra(a, bb[t], acc[t]);
  }

#pragma unroll
  for (int t = 0; t < 4; t++) {
    const int col = n0 + t * 16 + l16;
#pragma unroll
    for (int r = 0; r < 8; r++) {
      const int row = row0 + wave * 16 + r + 8 * lh;
      Out[(size_t)row * En + col] = acc[t][r] + bp[col];
    }
  }
}

// ---------------------------------------------------------------------------
// Launch
// ---------------------------------------------------------------------------
extern "C" void kernel_launch(void* const* d_in, const int* in_sizes, int n_in,
                              void* d_out, int out_size, void* d_ws,
                              size_t ws_size, hipStream_t stream) {
  const float* x  = (const float*)d_in[0];
  const float* Wq = (const float*)d_in[1];
  const float* Wk = (const float*)d_in[2];
  const float* Wv = (const float*)d_in[3];
  const float* Wp = (const float*)d_in[4];
  const float* bp = (const float*)d_in[5];
  float* out = (float*)d_out;

  char* p = (char*)d_ws;
  bf16* xb  = (bf16*)p; p += (size_t)BS * En * 2;
  bf16* wqb = (bf16*)p; p += (size_t)Hn * En * Dn * 2;
  bf16* wkb = (bf16*)p; p += (size_t)Hn * En * Dn * 2;
  bf16* wvb = (bf16*)p; p += (size_t)Hn * En * Dn * 2;
  bf16* wpb = (bf16*)p; p += (size_t)En * En * 2;
  bf16* Qb  = (bf16*)p; p += (size_t)Bn * Hn * Sn * Dn * 2;
  bf16* Kb  = (bf16*)p; p += (size_t)Bn * Hn * Sn * Dn * 2;
  bf16* Vb  = (bf16*)p; p += (size_t)Bn * Hn * Sn * Dn * 2;
  bf16* Cb  = (bf16*)p; p += (size_t)BS * En * 2;

  const int nx = BS * En, nw = Hn * En * Dn, np = En * En;
  cvt_k<<<(nx / 8 + 255) / 256, 256, 0, stream>>>(x, xb, nx);
  cvt_k<<<(nw / 8 + 255) / 256, 256, 0, stream>>>(Wq, wqb, nw);
  cvt_k<<<(nw / 8 + 255) / 256, 256, 0, stream>>>(Wk, wkb, nw);
  cvt_k<<<(nw / 8 + 255) / 256, 256, 0, stream>>>(Wv, wvb, nw);
  cvt_k<<<(np / 8 + 255) / 256, 256, 0, stream>>>(Wp, wpb, np);

  qkv_gemm_k<<<dim3(BS / 128, 3 * Hn), 256, 0, stream>>>(xb, wqb, wkb, wvb,
                                                         Qb, Kb, Vb);
  attn_k<<<dim3(Sn / 128, Hn, Bn), 256, 0, stream>>>(Qb, Kb, Vb, Cb);
  proj_k<<<dim3(BS / 128, En / 64), 256, 0, stream>>>(Cb, wpb, bp, out);
}
